// Mamba2Block_7352984011411
// MI455X (gfx1250) — compile-verified, matrix operand order checked by reading, not run on hardware
//
#include <hip/hip_runtime.h>
#include <hip/hip_bf16.h>

// ---------------- problem constants (from reference) ----------------
#define TT        8192          // total tokens = sum(PTS)
#define DIMX      256
#define D_INNER   1024
#define NHEADS    8
#define HEADDIM   128
#define D_STATE   128
#define CONV_DIM  1280          // D_INNER + 2*D_STATE
#define D_IN_PROJ 2312          // 2*D_INNER + 2*D_STATE + NHEADS
#define NSEQ      4

typedef __attribute__((ext_vector_type(16))) __bf16 v16bf;
typedef __attribute__((ext_vector_type(8)))  __bf16 v8bf;
typedef __attribute__((ext_vector_type(8)))  float  v8f;

__device__ __forceinline__ __bf16 f2bf(float f) {
    __hip_bfloat16 h = __float2bfloat16(f);
    __bf16 r;
    __builtin_memcpy(&r, &h, sizeof(r));
    return r;
}

__device__ __forceinline__ float silu_f(float v) {
    return v / (1.f + __expf(-v));
}

// Low 32 bits of a flat shared-memory address = LDS byte offset
// (ISA 10.2: the upper address bits are discarded).
__device__ __forceinline__ unsigned lds_off32(const void* p) {
    return (unsigned)(uintptr_t)p;
}

// Load 8 contiguous bf16 (16B) into fragment slots [o, o+8)
__device__ __forceinline__ void load8(v16bf& d, int o, const __bf16* p) {
    v8bf v = *reinterpret_cast<const v8bf*>(p);
#pragma unroll
    for (int i = 0; i < 8; ++i) d[o + i] = v[i];
}

// One wave computes a 16x16 f32 tile of  A[M,K] @ W[N,K]^T  via v_wmma_f32_16x16x32_bf16.
// ISA fragment layout (wave32):
//   A 16x32 bf16 : lane L -> M = L&15 ; elems 0..7 -> K = kh..kh+7, 8..15 -> K = 16+kh..  (kh = 8*(L>>4))
//   B 32x16 bf16 : mirrored with N = L&15 (W is row-major [N][K], rows are K-contiguous)
//   C/D 16x16 f32: VGPR r, lane L -> M = r + 8*(L>>4), N = L&15
__device__ v8f wmma_tile_bf16(const __bf16* __restrict__ A, int lda,
                              const __bf16* __restrict__ W, int ldw,
                              int K, int m0, int n0, int Nmax) {
    const int lane = threadIdx.x & 31;
    const int mr = m0 + (lane & 15);
    int nr = n0 + (lane & 15);
    if (nr >= Nmax) nr = Nmax - 1;            // clamp OOB partial tile (store guarded)
    const int kh = (lane >> 4) * 8;
    const __bf16* arow = A + (size_t)mr * lda + kh;
    const __bf16* brow = W + (size_t)nr * ldw + kh;
    v8f acc = {};
    for (int k0 = 0; k0 < K; k0 += 32) {
        if (k0 + 32 < K) {                    // gfx1250 global_prefetch_b8
            __builtin_prefetch(arow + k0 + 32, 0, 1);
            __builtin_prefetch(brow + k0 + 32, 0, 1);
        }
        v16bf a, b;
        load8(a, 0, arow + k0);
        load8(a, 8, arow + k0 + 16);
        load8(b, 0, brow + k0);
        load8(b, 8, brow + k0 + 16);
        acc = __builtin_amdgcn_wmma_f32_16x16x32_bf16(false, a, false, b,
                                                      (short)0, acc, false, false);
    }
    return acc;
}

// ---------------- stage 0: seq_idx / rev_idx ----------------
__global__ void seq_kernel(const int* __restrict__ pts,
                           int* __restrict__ seq, int* __restrict__ rev) {
    __shared__ int cu[NSEQ + 1];
    if (threadIdx.x == 0) {
        cu[0] = 0;
        for (int i = 0; i < NSEQ; ++i) cu[i + 1] = cu[i] + pts[i];
    }
    __syncthreads();
    for (int t = threadIdx.x; t < TT; t += blockDim.x) {
        int b = 0;
        while (b < NSEQ - 1 && t >= cu[b + 1]) ++b;
        seq[t] = b;
        rev[t] = cu[b] + cu[b + 1] - 1 - t;
    }
}

// ---------------- stage 1: LayerNorm -> u2 (fwd + reversed), res passthrough ----------------
__global__ void ln_kernel(const float* __restrict__ x,
                          const float* __restrict__ lw, const float* __restrict__ lb,
                          const int* __restrict__ rev,
                          __bf16* __restrict__ u2, float* __restrict__ res) {
    const int t = blockIdx.x, c = threadIdx.x;
    float v = x[(size_t)t * DIMX + c];
    __shared__ float s1[DIMX], s2[DIMX];
    s1[c] = v; s2[c] = v * v;
    __syncthreads();
    for (int s = DIMX / 2; s > 0; s >>= 1) {
        if (c < s) { s1[c] += s1[c + s]; s2[c] += s2[c + s]; }
        __syncthreads();
    }
    float mu  = s1[0] * (1.f / DIMX);
    float var = s2[0] * (1.f / DIMX) - mu * mu;
    float xn = (v - mu) * rsqrtf(var + 1e-5f) * lw[c] + lb[c];
    __bf16 bx = f2bf(xn);
    u2[(size_t)t * DIMX + c] = bx;                       // forward stream
    u2[((size_t)TT + rev[t]) * DIMX + c] = bx;           // reversed stream
    res[(size_t)t * DIMX + c] = v;                       // tuple output #2
}

// ---------------- stage 2: weights -> bf16 ----------------
__global__ void cast_kernel(const float* __restrict__ Win,  __bf16* __restrict__ bWin,
                            const float* __restrict__ Wout, __bf16* __restrict__ bWout,
                            const float* __restrict__ Wprj, __bf16* __restrict__ bWprj) {
    const int i = blockIdx.x * blockDim.x + threadIdx.x;
    const int stride = gridDim.x * blockDim.x;
    for (int j = i; j < D_IN_PROJ * DIMX;  j += stride) bWin[j]  = f2bf(Win[j]);
    for (int j = i; j < DIMX * D_INNER;    j += stride) bWout[j] = f2bf(Wout[j]);
    for (int j = i; j < DIMX * 2 * DIMX;   j += stride) bWprj[j] = f2bf(Wprj[j]);
}

// ---------------- stage 3: in-projection GEMM (WMMA) ----------------
__global__ void gemm_in_kernel(const __bf16* __restrict__ u2,
                               const __bf16* __restrict__ bWin,
                               float* __restrict__ zx) {
    const int wave = threadIdx.x >> 5;
    const int m0 = (blockIdx.y * 8 + wave) * 16;
    const int n0 = blockIdx.x * 16;
    v8f acc = wmma_tile_bf16(u2, DIMX, bWin, DIMX, DIMX, m0, n0, D_IN_PROJ);
    const int lane = threadIdx.x & 31;
    const int n = n0 + (lane & 15);
    if (n < D_IN_PROJ) {
#pragma unroll
        for (int r = 0; r < 8; ++r) {
            int m = m0 + r + (lane >> 4) * 8;
            zx[(size_t)m * D_IN_PROJ + n] = acc[r];
        }
    }
}

// ---------------- stage 4: causal depthwise conv + silu, softplus(dt) ----------------
__global__ void conv_kernel(const float* __restrict__ zx, const int* __restrict__ seq,
                            const float* __restrict__ cw, const float* __restrict__ cb,
                            const float* __restrict__ dtb,
                            float* __restrict__ xbc, float* __restrict__ dtg) {
    const int blk = blockIdx.x;
    const int dir = blk >> 13, t = blk & (TT - 1);
    const size_t row = (size_t)dir * TT + t;
    const int st = seq[t];
    for (int c = threadIdx.x; c < CONV_DIM; c += blockDim.x) {
        float acc = cb[c];
#pragma unroll
        for (int j = 0; j < 4; ++j) {
            int tt = t - 3 + j;
            if (tt >= 0 && seq[tt] == st)
                acc += cw[c * 4 + j] * zx[((size_t)dir * TT + tt) * D_IN_PROJ + D_INNER + c];
        }
        xbc[row * CONV_DIM + c] = silu_f(acc);
    }
    if (threadIdx.x < NHEADS) {
        int h = threadIdx.x;
        float v = zx[row * D_IN_PROJ + (D_IN_PROJ - NHEADS) + h] + dtb[h];
        dtg[row * NHEADS + h] = (v > 20.f) ? v : log1pf(__expf(v));
    }
}

// ---------------- stage 5: selective scan ----------------
// Serial critical path. B/C/x for each timestep are streamed global->LDS with
// CDNA5 async-to-LDS ops (no VGPR staging), double-buffered two steps deep so
// HBM latency hides behind one full compute iteration. Each wave issues
// exactly 2 async b32 instructions per step (uniform ASYNCcnt bookkeeping):
//   instr 1: word tid        -> B (0..127) and C (128..255)
//   instr 2: word 256+tid&127-> x (duplicated by upper half-block, benign)
#define STG_WORDS 384

__device__ __forceinline__ void scan_issue_async(const float* __restrict__ row, int head,
                                                 float* __restrict__ dst, int tid) {
    {
        unsigned l = lds_off32(dst + tid);
        const float* g = row + D_INNER + tid;                 // B | C words
        asm volatile("global_load_async_to_lds_b32 %0, %1, off"
                     :: "v"(l), "v"(g) : "memory");
    }
    {
        int w = tid & (HEADDIM - 1);
        unsigned l = lds_off32(dst + 256 + w);
        const float* g = row + head * HEADDIM + w;            // x words
        asm volatile("global_load_async_to_lds_b32 %0, %1, off"
                     :: "v"(l), "v"(g) : "memory");
    }
}

__global__ void __launch_bounds__(256)
scan_kernel(const float* __restrict__ xbc, const float* __restrict__ dtg,
            const int* __restrict__ seq, const float* __restrict__ A_log,
            const float* __restrict__ Dp, float* __restrict__ ys) {
    const int dir = blockIdx.x >> 3, head = blockIdx.x & 7;
    const int tid = threadIdx.x;
    const int p = tid >> 1;            // 0..127  (HEADDIM row)
    const int half = tid & 1;          // which 64-wide n slice
    __shared__ float stage[2][STG_WORDS];
    float h[64];
#pragma unroll
    for (int i = 0; i < 64; ++i) h[i] = 0.f;
    const float A  = -__expf(A_log[head]);
    const float Dv = Dp[head];
    const size_t base = (size_t)dir * TT;

    // prime the 2-deep pipeline
    scan_issue_async(xbc + (base + 0) * CONV_DIM, head, stage[0], tid);
    scan_issue_async(xbc + (base + 1) * CONV_DIM, head, stage[1], tid);

    int prev = -1;
    for (int t = 0; t < TT; ++t) {
        // retire group t (leave group t+1 in flight), then make it WG-visible
        if (t < TT - 1) asm volatile("s_wait_asynccnt 0x2" ::: "memory");
        else            asm volatile("s_wait_asynccnt 0x0" ::: "memory");
        __syncthreads();

        const float* st = stage[t & 1];
        const int sq = seq[t];
        const float keep = (t > 0 && sq == prev) ? 1.f : 0.f;
        prev = sq;
        const float dtv  = dtg[(base + t) * NHEADS + head];
        const float xp   = st[256 + p];
        const float dtx  = dtv * xp;
        const float coef = keep * __expf(dtv * A);
        const float* Bh = st + half * 64;
        const float* Ch = st + 128 + half * 64;
        float acc = 0.f;
#pragma unroll 8
        for (int i = 0; i < 64; ++i) {
            h[i] = coef * h[i] + dtx * Bh[i];
            acc += h[i] * Ch[i];
        }
        acc += __shfl_xor(acc, 1, 32); // combine the two n-halves (adjacent lanes)
        if (half == 0)
            ys[(base + t) * D_INNER + head * HEADDIM + p] = acc + Dv * xp;

        __syncthreads();               // everyone done reading stage[t&1]
        if (t + 2 < TT)                // refill it for step t+2
            scan_issue_async(xbc + (base + t + 2) * CONV_DIM, head, stage[t & 1], tid);
    }
}

// ---------------- stage 6: gate with silu(z), grouped RMS-norm -> bf16 ----------------
__global__ void gate_norm_kernel(const float* __restrict__ ys, const float* __restrict__ zx,
                                 const float* __restrict__ gnw, __bf16* __restrict__ ygn) {
    const int blk = blockIdx.x;
    const size_t r = (size_t)(blk >> 13) * TT + (blk & (TT - 1));
    __shared__ float red[256];
    float g[4]; float ss = 0.f;
#pragma unroll
    for (int i = 0; i < 4; ++i) {
        int c = threadIdx.x * 4 + i;
        float z = zx[r * D_IN_PROJ + c];
        float v = ys[r * D_INNER + c] * silu_f(z);
        g[i] = v; ss += v * v;
    }
    red[threadIdx.x] = ss;
    __syncthreads();
    for (int s = 128; s > 0; s >>= 1) {
        if (threadIdx.x < s) red[threadIdx.x] += red[threadIdx.x + s];
        __syncthreads();
    }
    const float rms = rsqrtf(red[0] * (1.f / D_INNER) + 1e-5f);
#pragma unroll
    for (int i = 0; i < 4; ++i) {
        int c = threadIdx.x * 4 + i;
        ygn[r * D_INNER + c] = f2bf(g[i] * rms * gnw[c]);
    }
}

// ---------------- stage 7: out-projection GEMM (WMMA) + direction-merge into cat ----------------
__global__ void gemm_out_kernel(const __bf16* __restrict__ ygn, const __bf16* __restrict__ bWout,
                                const int* __restrict__ rev, __bf16* __restrict__ cat) {
    const int wave = threadIdx.x >> 5;
    const int m0 = (blockIdx.y * 8 + wave) * 16;
    const int n0 = blockIdx.x * 16;
    v8f acc = wmma_tile_bf16(ygn, D_INNER, bWout, D_INNER, D_INNER, m0, n0, DIMX);
    const int lane = threadIdx.x & 31;
    const int n = n0 + (lane & 15);
#pragma unroll
    for (int r = 0; r < 8; ++r) {
        int m = m0 + r + (lane >> 4) * 8;
        int dir = m >> 13, t = m & (TT - 1);
        int orow = dir ? rev[t] : t;             // un-reverse backward direction
        int ocol = dir ? (DIMX + n) : n;
        cat[(size_t)orow * (2 * DIMX) + ocol] = f2bf(acc[r]);
    }
}

// ---------------- stage 8: final projection GEMM (WMMA) + bias -> d_out ----------------
__global__ void gemm_proj_kernel(const __bf16* __restrict__ cat, const __bf16* __restrict__ bWprj,
                                 const float* __restrict__ bproj, float* __restrict__ out) {
    const int wave = threadIdx.x >> 5;
    const int m0 = (blockIdx.y * 8 + wave) * 16;
    const int n0 = blockIdx.x * 16;
    v8f acc = wmma_tile_bf16(cat, 2 * DIMX, bWprj, 2 * DIMX, 2 * DIMX, m0, n0, DIMX);
    const int lane = threadIdx.x & 31;
    const int n = n0 + (lane & 15);
    const float bn = bproj[n];
#pragma unroll
    for (int r = 0; r < 8; ++r) {
        int m = m0 + r + (lane >> 4) * 8;
        out[(size_t)m * DIMX + n] = acc[r] + bn;
    }
}

// ---------------- host launcher ----------------
extern "C" void kernel_launch(void* const* d_in, const int* in_sizes, int n_in,
                              void* d_out, int out_size, void* d_ws, size_t ws_size,
                              hipStream_t stream) {
    const float* x       = (const float*)d_in[0];
    const int*   pts     = (const int*)  d_in[1];
    const float* ln_w    = (const float*)d_in[2];
    const float* ln_b    = (const float*)d_in[3];
    const float* W_in    = (const float*)d_in[4];
    const float* conv_w  = (const float*)d_in[5];
    const float* conv_b  = (const float*)d_in[6];
    const float* dt_bias = (const float*)d_in[7];
    const float* A_log   = (const float*)d_in[8];
    const float* Dp      = (const float*)d_in[9];
    const float* gn_w    = (const float*)d_in[10];
    const float* W_out   = (const float*)d_in[11];
    const float* W_proj  = (const float*)d_in[12];
    const float* b_proj  = (const float*)d_in[13];
    float* out = (float*)d_out;                  // [T,256] out then [T,256] res

    char* ws = (char*)d_ws;
    size_t off = 0;
    auto take = [&](size_t bytes) -> void* {
        void* p = ws + off;
        off = (off + bytes + 255) & ~(size_t)255;
        return p;
    };
    int*    seq   = (int*)   take((size_t)TT * 4);
    int*    rev   = (int*)   take((size_t)TT * 4);
    __bf16* u2    = (__bf16*)take((size_t)2 * TT * DIMX * 2);
    __bf16* bWin  = (__bf16*)take((size_t)D_IN_PROJ * DIMX * 2);
    __bf16* bWout = (__bf16*)take((size_t)DIMX * D_INNER * 2);
    __bf16* bWprj = (__bf16*)take((size_t)DIMX * 2 * DIMX * 2);
    float*  zx    = (float*) take((size_t)2 * TT * D_IN_PROJ * 4);
    float*  xbc   = (float*) take((size_t)2 * TT * CONV_DIM * 4);
    float*  dtg   = (float*) take((size_t)2 * TT * NHEADS * 4);
    float*  ys    = (float*) take((size_t)2 * TT * D_INNER * 4);
    __bf16* ygn   = (__bf16*)take((size_t)2 * TT * D_INNER * 2);
    __bf16* cat   = (__bf16*)take((size_t)TT * 2 * DIMX * 2);
    (void)ws_size; (void)in_sizes; (void)n_in; (void)out_size;

    seq_kernel<<<1, 256, 0, stream>>>(pts, seq, rev);
    ln_kernel<<<TT, 256, 0, stream>>>(x, ln_w, ln_b, rev, u2, out + (size_t)TT * DIMX);
    cast_kernel<<<2048, 256, 0, stream>>>(W_in, bWin, W_out, bWout, W_proj, bWprj);

    {   // in-proj: M=2T=16384 (1024 m-tiles, 8 per block), N=2312 -> 145 n-tiles
        dim3 grid((D_IN_PROJ + 15) / 16, (2 * TT / 16) / 8);
        gemm_in_kernel<<<grid, 256, 0, stream>>>(u2, bWin, zx);
    }
    conv_kernel<<<2 * TT, 256, 0, stream>>>(zx, seq, conv_w, conv_b, dt_bias, xbc, dtg);
    scan_kernel<<<2 * NHEADS, 256, 0, stream>>>(xbc, dtg, seq, A_log, Dp, ys);
    gate_norm_kernel<<<2 * TT, 256, 0, stream>>>(ys, zx, gn_w, ygn);
    {   // out-proj: M=16384, N=256
        dim3 grid(DIMX / 16, (2 * TT / 16) / 8);
        gemm_out_kernel<<<grid, 256, 0, stream>>>(ygn, bWout, rev, cat);
    }
    {   // final proj: M=8192, N=256
        dim3 grid(DIMX / 16, (TT / 16) / 8);
        gemm_proj_kernel<<<grid, 256, 0, stream>>>(cat, bWprj, b_proj, out);
    }
}
